// Attention_47562467836169
// MI455X (gfx1250) — matrix operand order repaired and checked by reading, not compiled, not run on hardware
//
#include <hip/hip_runtime.h>

typedef __attribute__((ext_vector_type(16))) _Float16 v16h;
typedef __attribute__((ext_vector_type(8)))  float    v8f;

constexpr int S_LEN   = 4096;
constexpr int BATCH   = 128;
constexpr int H_DIM   = 64;
constexpr int THREADS = 512;          // 16 waves (wave32)
constexpr int WAVES   = THREADS / 32;
constexpr int NTILES  = S_LEN / 16;   // 256 row-tiles of 16 seq positions

__device__ __forceinline__ float fast_tanh(float x) {
    x = fminf(fmaxf(x, -9.0f), 9.0f);
    const float e = __builtin_amdgcn_exp2f(x * 2.88539008177793f); // exp(2x)
    return (e - 1.0f) * __builtin_amdgcn_rcpf(e + 1.0f);
}

__device__ __forceinline__ void pack4(v16h& a, int base, float4 c) {
    a[base + 0] = (_Float16)c.x;
    a[base + 1] = (_Float16)c.y;
    a[base + 2] = (_Float16)c.z;
    a[base + 3] = (_Float16)c.w;
}

__global__ __launch_bounds__(THREADS) void additive_attn_kernel(
    const float* __restrict__ enc,   // [S,B,H]
    const float* __restrict__ dec,   // [1,B,H]
    const float* __restrict__ Wq_w,  // [H,H]
    const float* __restrict__ Wq_b,  // [H]
    const float* __restrict__ Wk_w,  // [H,H]
    const float* __restrict__ Wk_b,  // [H]
    const float* __restrict__ W_w,   // [1,H]
    const float* __restrict__ W_b,   // [1]
    float* __restrict__ out)         // [1,B,H]
{
    __shared__ float q_lds[H_DIM];
    __shared__ float sc[S_LEN];
    __shared__ float red[THREADS];
    __shared__ float ctx[16 * H_DIM];

    const int b    = blockIdx.x;
    const int tid  = threadIdx.x;
    const int wave = tid >> 5;
    const int lane = tid & 31;
    const int hl   = lane & 15;
    const bool hiHalf = lane >= 16;

    // ---- q = dec @ Wq_w^T + Wq_b  (tiny: 64 dot products) ----
    if (tid < H_DIM) {
        float a = Wq_b[tid];
        const float* wr = Wq_w + tid * H_DIM;
        const float* dv = dec  + b  * H_DIM;
        for (int j = 0; j < H_DIM; ++j) a = fmaf(wr[j], dv[j], a);
        q_lds[tid] = a;
    }

    // ---- resident B fragments for Wk^T: B[j = kk*32 + 8*(lane/16) + (t<8 ? t : t+8)][h = nt*16+(lane&15)] ----
    v16h bfrag[4][2];
#pragma unroll
    for (int nt = 0; nt < 4; ++nt)
#pragma unroll
        for (int kk = 0; kk < 2; ++kk)
#pragma unroll
            for (int t = 0; t < 16; ++t)
                bfrag[nt][kk][t] = (_Float16)Wk_w[(nt * 16 + hl) * H_DIM + kk * 32 + (hiHalf ? 8 : 0) + (t < 8 ? t : t + 8)];

    float ww[4], wkb[4];
#pragma unroll
    for (int nt = 0; nt < 4; ++nt) {
        const int h = nt * 16 + hl;
        ww[nt]  = W_w[h];
        wkb[nt] = Wk_b[h];
    }
    const float Wb = W_b[0];

    __syncthreads();

    float qv[4];
#pragma unroll
    for (int nt = 0; nt < 4; ++nt) qv[nt] = q_lds[nt * 16 + hl] + wkb[nt];

    // ---- fused k-GEMM + tanh + score, 16 seq rows per tile ----
    // A-fragment gather: lane L owns row (L&15); 32B column chunks at jb ∈ {0,8}.
    // Across the wave these 32B chunks exactly tile the 16 rows' cachelines.
    for (int tile = wave; tile < NTILES; tile += WAVES) {
        const int sbase = tile * 16;
        const int rbase = ((sbase + hl) * BATCH + b) * H_DIM + (hiHalf ? 8 : 0);

        const float4 c0 = *reinterpret_cast<const float4*>(enc + rbase);
        const float4 c1 = *reinterpret_cast<const float4*>(enc + rbase + 4);
        const float4 c2 = *reinterpret_cast<const float4*>(enc + rbase + 16);
        const float4 c3 = *reinterpret_cast<const float4*>(enc + rbase + 20);
        const float4 c4 = *reinterpret_cast<const float4*>(enc + rbase + 32);
        const float4 c5 = *reinterpret_cast<const float4*>(enc + rbase + 36);
        const float4 c6 = *reinterpret_cast<const float4*>(enc + rbase + 48);
        const float4 c7 = *reinterpret_cast<const float4*>(enc + rbase + 52);

        if (tile + WAVES < NTILES) { // prefetch next tile's lines for this lane
            const int nbase = rbase + 16 * WAVES * BATCH * H_DIM;
            __builtin_prefetch(enc + nbase,      0, 1);
            __builtin_prefetch(enc + nbase + 32, 0, 1);
        }

        v16h a0, a1;
        pack4(a0, 0, c0);  pack4(a0, 4, c1);  pack4(a0, 8, c2);  pack4(a0, 12, c3);
        pack4(a1, 0, c4);  pack4(a1, 4, c5);  pack4(a1, 8, c6);  pack4(a1, 12, c7);

        // k[16x64] = A[16x64] x Wk^T[64x64] via 8 WMMAs (f32 accum)
        v8f acc[4];
        const v8f czero = {};
#pragma unroll
        for (int nt = 0; nt < 4; ++nt) {
            acc[nt] = __builtin_amdgcn_wmma_f32_16x16x32_f16(
                false, a0, false, bfrag[nt][0], (short)0, czero, false, false);
            acc[nt] = __builtin_amdgcn_wmma_f32_16x16x32_f16(
                false, a1, false, bfrag[nt][1], (short)0, acc[nt], false, false);
        }

        // score[s] = W_w . tanh(q + k_row + Wk_b) + W_b
#pragma unroll
        for (int r = 0; r < 8; ++r) {
            float p = 0.0f;
#pragma unroll
            for (int nt = 0; nt < 4; ++nt)
                p = fmaf(ww[nt], fast_tanh(acc[nt][r] + qv[nt]), p);
#pragma unroll
            for (int off = 8; off > 0; off >>= 1)
                p += __shfl_xor(p, off, 16);
            if (hl == 0)
                sc[sbase + r + (hiHalf ? 8 : 0)] = p + Wb;
        }
    }
    __syncthreads();

    // ---- softmax over S in LDS ----
    constexpr int PER = S_LEN / THREADS; // 8
    float lm = -3.0e38f;
#pragma unroll
    for (int i = 0; i < PER; ++i) lm = fmaxf(lm, sc[tid * PER + i]);
    red[tid] = lm;
    __syncthreads();
    for (int off = THREADS / 2; off > 0; off >>= 1) {
        if (tid < off) red[tid] = fmaxf(red[tid], red[tid + off]);
        __syncthreads();
    }
    const float gmax = red[0];
    __syncthreads();

    float lsum = 0.0f;
#pragma unroll
    for (int i = 0; i < PER; ++i) {
        const float e = __builtin_amdgcn_exp2f((sc[tid * PER + i] - gmax) * 1.44269504f);
        sc[tid * PER + i] = e;
        lsum += e;
    }
    red[tid] = lsum;
    __syncthreads();
    for (int off = THREADS / 2; off > 0; off >>= 1) {
        if (tid < off) red[tid] += red[tid + off];
        __syncthreads();
    }
    const float inv = __builtin_amdgcn_rcpf(red[0]);
    __syncthreads();

    // ---- context = attn @ enc : 16 groups x 32 lanes, float2 per lane ----
    // Each group of 32 lanes covers one full 256B enc row per iteration.
    const int hp = (tid & 31) * 2;
    const int g  = tid >> 5; // 0..15
    float cx = 0.0f, cy = 0.0f;
    for (int s = g; s < S_LEN; s += 16) {
        const float2 e = *reinterpret_cast<const float2*>(
            enc + (s * BATCH + b) * H_DIM + hp);
        const float a = sc[s];
        cx = fmaf(a, e.x, cx);
        cy = fmaf(a, e.y, cy);
        if (s + 16 < S_LEN)
            __builtin_prefetch(enc + ((s + 16) * BATCH + b) * H_DIM + hp, 0, 1);
    }
    ctx[g * H_DIM + hp]     = cx;
    ctx[g * H_DIM + hp + 1] = cy;
    __syncthreads();
    if (tid < H_DIM) {
        float t = 0.0f;
#pragma unroll
        for (int gg = 0; gg < 16; ++gg) t += ctx[gg * H_DIM + tid];
        out[b * H_DIM + tid] = t * inv;
    }
}

extern "C" void kernel_launch(void* const* d_in, const int* in_sizes, int n_in,
                              void* d_out, int out_size, void* d_ws, size_t ws_size,
                              hipStream_t stream) {
    (void)in_sizes; (void)n_in; (void)out_size; (void)d_ws; (void)ws_size;
    const float* enc  = (const float*)d_in[0];
    const float* dec  = (const float*)d_in[1];
    const float* Wq_w = (const float*)d_in[2];
    const float* Wq_b = (const float*)d_in[3];
    const float* Wk_w = (const float*)d_in[4];
    const float* Wk_b = (const float*)d_in[5];
    const float* W_w  = (const float*)d_in[6];
    const float* W_b  = (const float*)d_in[7];
    float* out = (float*)d_out;

    additive_attn_kernel<<<BATCH, THREADS, 0, stream>>>(
        enc, dec, Wq_w, Wq_b, Wk_w, Wk_b, W_w, W_b, out);
}
